// C2QAttention_76897094468112
// MI455X (gfx1250) — matrix operand order repaired and checked by reading, not compiled, not run on hardware
//
#include <hip/hip_runtime.h>
#include <hip/hip_bf16.h>

#define B_  8
#define TC  2048
#define TQ  1024
#define DD  1024

typedef __attribute__((ext_vector_type(16))) __bf16 v16bf;
typedef __attribute__((ext_vector_type(8)))  float  v8f;

union Frag16 { uint4 q[2]; v16bf v; };
union QU     { uint4 q; unsigned int u[4]; unsigned short s[8]; };

__device__ __forceinline__ unsigned short f32_bf16_rne(float f) {
  unsigned int u = __float_as_uint(f);
  u += 0x7FFFu + ((u >> 16) & 1u);
  return (unsigned short)(u >> 16);
}

// ---------------------------------------------------------------------------
// Kernel 1: row softmax over Tq, one wave32 per row, bf16 output.
// ---------------------------------------------------------------------------
__global__ __launch_bounds__(256) void softmax_rows(const float* __restrict__ sim,
                                                    unsigned short* __restrict__ attnb) {
  const int wave = threadIdx.x >> 5;
  const int lane = threadIdx.x & 31;
  const int row  = blockIdx.x * 8 + wave;                 // [0, B*TC)
  const float* src = sim + (size_t)row * TQ;

  float4 v[8];
  float m = -3.4e38f;
#pragma unroll
  for (int c = 0; c < 8; ++c) {
    v[c] = ((const float4*)src)[lane + c * 32];
    m = fmaxf(m, fmaxf(fmaxf(v[c].x, v[c].y), fmaxf(v[c].z, v[c].w)));
  }
#pragma unroll
  for (int off = 16; off > 0; off >>= 1) m = fmaxf(m, __shfl_xor(m, off, 32));

  float s = 0.f;
#pragma unroll
  for (int c = 0; c < 8; ++c) {
    v[c].x = __expf(v[c].x - m);
    v[c].y = __expf(v[c].y - m);
    v[c].z = __expf(v[c].z - m);
    v[c].w = __expf(v[c].w - m);
    s += (v[c].x + v[c].y) + (v[c].z + v[c].w);
  }
#pragma unroll
  for (int off = 16; off > 0; off >>= 1) s += __shfl_xor(s, off, 32);
  const float inv = 1.0f / s;

  unsigned short* dst = attnb + (size_t)row * TQ;
#pragma unroll
  for (int c = 0; c < 8; ++c) {
    uint2 p;
    p.x = (unsigned int)f32_bf16_rne(v[c].x * inv) |
          ((unsigned int)f32_bf16_rne(v[c].y * inv) << 16);
    p.y = (unsigned int)f32_bf16_rne(v[c].z * inv) |
          ((unsigned int)f32_bf16_rne(v[c].w * inv) << 16);
    ((uint2*)dst)[lane + c * 32] = p;
  }
}

// ---------------------------------------------------------------------------
// Kernel 2: f32 -> bf16 bulk convert (qencode).
// ---------------------------------------------------------------------------
__global__ __launch_bounds__(256) void cvt_bf16(const float* __restrict__ x,
                                                unsigned short* __restrict__ y) {
  const int i = blockIdx.x * 256 + threadIdx.x;
  float4 f = ((const float4*)x)[i];
  uint2 p;
  p.x = (unsigned int)f32_bf16_rne(f.x) | ((unsigned int)f32_bf16_rne(f.y) << 16);
  p.y = (unsigned int)f32_bf16_rne(f.z) | ((unsigned int)f32_bf16_rne(f.w) << 16);
  ((uint2*)y)[i] = p;
}

// ---------------------------------------------------------------------------
// Kernel 3: batched GEMM  out[b] = attn[b] (2048x1024) @ q[b] (1024x1024)
// bf16 WMMA, f32 accumulate. Block tile 128x128, BK=32, 8 waves of 64x32.
// Double-buffered LDS via pointer ping-pong (single loop body, unroll 1);
// A tile staged with GLOBAL_LOAD_ASYNC_TO_LDS_B128; B transpose via v_perm_b32.
// ---------------------------------------------------------------------------
__global__ __launch_bounds__(256) void c2q_wmma(const unsigned short* __restrict__ attnb,
                                                const unsigned short* __restrict__ qb,
                                                float* __restrict__ out) {
  __shared__ unsigned short As[2][128 * 32];   // [m][k] row-major
  __shared__ unsigned short Bs[2][128 * 32];   // [n][k] (B transposed)

  const int batch = blockIdx.z;
  const int bm    = blockIdx.y * 128;
  const int bn    = blockIdx.x * 128;
  const int t     = threadIdx.x;
  const int wave  = t >> 5;
  const int lane  = t & 31;
  const int waveM = wave >> 2;              // 0..1  -> 64 rows each
  const int waveN = wave & 3;               // 0..3  -> 32 cols each
  const int laneLo = lane & 15;
  const int laneHi = lane >> 4;             // 0 or 1 (K-half select)

  const unsigned short* Aglob = attnb + (size_t)batch * TC * TQ;
  const unsigned short* Bglob = qb    + (size_t)batch * TQ * DD;

  // Global->LDS assignments
  const int arow = t >> 1;                  // A: 2 threads per row, 16 elems each
  const int akk  = (t & 1) * 16;
  const int bn8  = (t & 15) * 8;            // B: 8(N) x 2(K) micro-tile per thread
  const int bk2  = (t >> 4) * 2;

  const v8f vzero = {};
  v8f acc[4][2];
#pragma unroll
  for (int m = 0; m < 4; ++m)
#pragma unroll
    for (int n = 0; n < 2; ++n) acc[m][n] = vzero;

  // Async copy of this thread's 32-byte A slice into the given LDS buffer.
  // IOFFSET applies to both LDS and global addresses (ISA 08, §4.4).
  auto async_copy_A = [&](unsigned short* dstBase, int k0) {
    unsigned lds = (unsigned)(size_t)(dstBase + arow * 32 + akk);
    unsigned long long ga =
        (unsigned long long)(size_t)(Aglob + (size_t)(bm + arow) * TQ + (k0 + akk));
    asm volatile("global_load_async_to_lds_b128 %0, %1, off\n\t"
                 "global_load_async_to_lds_b128 %0, %1, off offset:16"
                 :: "v"(lds), "v"(ga) : "memory");
  };

  // Transpose-pack two K-rows (8 N each) with v_perm_b32:
  // out j = { rowK1.half(j), rowK0.half(j) }.
  auto store_B = [&](unsigned short* base, const QU& r0, const QU& r1) {
#pragma unroll
    for (int j = 0; j < 8; ++j) {
      unsigned pk = __builtin_amdgcn_perm(r1.u[j >> 1], r0.u[j >> 1],
                                          (j & 1) ? 0x07060302u : 0x05040100u);
      *(unsigned*)&base[(bn8 + j) * 32 + bk2] = pk;
    }
  };

  auto compute_tile = [&](const unsigned short* Asb, const unsigned short* Bsb) {
    Frag16 afr[4], bfr[2];
    const int aK = laneHi * 8;              // lanes 0-15: K0..7/16..23; 16-31: K8..15/24..31
#pragma unroll
    for (int m = 0; m < 4; ++m) {
      const unsigned short* p = &Asb[(waveM * 64 + m * 16 + laneLo) * 32 + aK];
      afr[m].q[0] = *(const uint4*)p;
      afr[m].q[1] = *(const uint4*)(p + 16);
    }
#pragma unroll
    for (int n = 0; n < 2; ++n) {
      const unsigned short* p = &Bsb[(waveN * 32 + n * 16 + laneLo) * 32 + laneHi * 8];
      bfr[n].q[0] = *(const uint4*)p;
      bfr[n].q[1] = *(const uint4*)(p + 16);
    }
#pragma unroll
    for (int m = 0; m < 4; ++m)
#pragma unroll
      for (int n = 0; n < 2; ++n)
        acc[m][n] = __builtin_amdgcn_wmma_f32_16x16x32_bf16(
            false, afr[m].v, false, bfr[n].v, (short)0, acc[m][n], false, false);
  };

  // ---- prologue: stage tile 0 into buffer 0 ----
  {
    const unsigned short* bp = Bglob + (size_t)bk2 * DD + (bn + bn8);
    QU b0, b1;
    b0.q = *(const uint4*)bp;
    b1.q = *(const uint4*)(bp + DD);
    store_B(Bs[0], b0, b1);
    async_copy_A(As[0], 0);
  }

  // ---- main pipeline: single body, pointer ping-pong ----
  unsigned short* Acur = As[0];
  unsigned short* Anxt = As[1];
  unsigned short* Bcur = Bs[0];
  unsigned short* Bnxt = Bs[1];
#pragma unroll 1
  for (int i = 0; i < TQ / 32 - 1; ++i) {
    asm volatile("s_wait_asynccnt 0x0" ::: "memory");  // tile i's async A writes done
    __syncthreads();                                   // current buffers visible to all

    const unsigned short* bp = Bglob + (size_t)((i + 1) * 32 + bk2) * DD + (bn + bn8);
    QU nb0, nb1;
    nb0.q = *(const uint4*)bp;
    nb1.q = *(const uint4*)(bp + DD);
    async_copy_A(Anxt, (i + 1) * 32);                  // overlap with compute below

    compute_tile(Acur, Bcur);                          // 8x v_wmma_f32_16x16x32_bf16

    store_B(Bnxt, nb0, nb1);

    unsigned short* tA = Acur; Acur = Anxt; Anxt = tA;
    unsigned short* tB = Bcur; Bcur = Bnxt; Bnxt = tB;
  }
  // ---- epilogue tile (no prefetch) ----
  asm volatile("s_wait_asynccnt 0x0" ::: "memory");
  __syncthreads();
  compute_tile(Acur, Bcur);

  // Epilogue: C layout -> VGPR r: lanes 0-15 M=r, lanes 16-31 M=r+8; N=lane&15
  float* O = out + (size_t)batch * TC * DD;
#pragma unroll
  for (int m = 0; m < 4; ++m) {
    const int rowBase = bm + waveM * 64 + m * 16 + laneHi * 8;
#pragma unroll
    for (int n = 0; n < 2; ++n) {
      const int col = bn + waveN * 32 + n * 16 + laneLo;
#pragma unroll
      for (int r = 0; r < 8; ++r)
        O[(size_t)(rowBase + r) * DD + col] = acc[m][n][r];
    }
  }
}

extern "C" void kernel_launch(void* const* d_in, const int* in_sizes, int n_in,
                              void* d_out, int out_size, void* d_ws, size_t ws_size,
                              hipStream_t stream) {
  (void)in_sizes; (void)n_in; (void)out_size; (void)ws_size;
  const float* sim = (const float*)d_in[0];   // [B, Tc, Tq] f32
  const float* qen = (const float*)d_in[1];   // [B, Tq, D]  f32
  float* out = (float*)d_out;                 // [B, Tc, D]  f32

  unsigned short* attnb = (unsigned short*)d_ws;                 // 32 MiB bf16
  unsigned short* qb    = attnb + (size_t)B_ * TC * TQ;          // 16 MiB bf16

  softmax_rows<<<dim3((B_ * TC) / 8), 256, 0, stream>>>(sim, attnb);
  cvt_bf16<<<dim3((B_ * TQ * DD / 4) / 256), 256, 0, stream>>>(qen, qb);

  dim3 grid(DD / 128, TC / 128, B_);
  c2q_wmma<<<grid, 256, 0, stream>>>(attnb, qb, out);
}
